// RNAStructurePredictor_24601572671731
// MI455X (gfx1250) — matrix operand order repaired and checked by reading, not compiled, not run on hardware
//
#include <hip/hip_runtime.h>
#include <math.h>

#define DF 128

typedef __attribute__((ext_vector_type(8)))  float  v8f;
typedef __attribute__((ext_vector_type(16))) __bf16 v16bf;

union BF16x16 { v16bf v; unsigned short u[16]; };

__device__ __forceinline__ unsigned short f2bf(float f) {
  unsigned int u = __float_as_uint(f);
  u += 0x7FFFu + ((u >> 16) & 1u);          // round-to-nearest-even
  return (unsigned short)(u >> 16);
}

// ---------------- weight fp32 -> bf16 (one 128x128 matrix) ----------------
__global__ void k_wconv(const float* __restrict__ w, unsigned short* __restrict__ out) {
  int i = blockIdx.x * blockDim.x + threadIdx.x;   // 64*256 == 16384
  out[i] = f2bf(w[i]);
}

// ---------------- zero fill ----------------
__global__ void k_zero(float* __restrict__ p, long long n) {
  long long i = (long long)blockIdx.x * blockDim.x + threadIdx.x;
  long long s = (long long)gridDim.x * blockDim.x;
  for (; i < n; i += s) p[i] = 0.0f;
}

// ---------------- degree count ----------------
__global__ void k_deg(const int* __restrict__ row, int E, float* __restrict__ deg) {
  int i = blockIdx.x * blockDim.x + threadIdx.x;
  int s = gridDim.x * blockDim.x;
  for (; i < E; i += s) atomicAdd(&deg[row[i]], 1.0f);
}

// ---------------- fused GEMM: C_j[N x 128] = X[N x 128] @ W_j^T + b_j  (WMMA bf16) ----
// NMAT weight matrices share one A-fragment build (Q/K/V of an attention layer).
// W stored bf16 row-major [out][in]; B element (k,o) = W[o*128+k] -> contiguous in k.
template <int LDC, int ACT, int NMAT>
__global__ void __launch_bounds__(256) k_gemm(const float* __restrict__ X,
    const unsigned short* __restrict__ Wb0, const float* __restrict__ b0, float* __restrict__ C0,
    const unsigned short* __restrict__ Wb1, const float* __restrict__ b1, float* __restrict__ C1,
    const unsigned short* __restrict__ Wb2, const float* __restrict__ b2, float* __restrict__ C2,
    int N) {
  const int lane = threadIdx.x & 31;
  const int wv   = threadIdx.x >> 5;
  const int rowBase = blockIdx.x * 128 + wv * 16;
  if (rowBase >= N) return;                 // wave-uniform: EXEC stays all-ones for WMMA
  const int  m  = lane & 15;
  const bool hi = lane >= 16;

  int rowA = rowBase + m; if (rowA > N - 1) rowA = N - 1;
  const float* xrow = X + (size_t)rowA * DF;

  // A fragments (16x32 bf16, this code's contiguous k order, same as B): lanes 0-15 hold K {0..15}, lanes 16-31 hold {16..31}
  BF16x16 a[4];
#pragma unroll
  for (int kt = 0; kt < 4; ++kt) {
    const float4* x4 = (const float4*)(xrow + kt * 32 + (hi ? 16 : 0));
    float4 p0 = x4[0], p1 = x4[1];   // k = base .. base+7
    float4 p2 = x4[2], p3 = x4[3];   // k = base+8 .. base+15
    a[kt].u[0]=f2bf(p0.x); a[kt].u[1]=f2bf(p0.y); a[kt].u[2]=f2bf(p0.z); a[kt].u[3]=f2bf(p0.w);
    a[kt].u[4]=f2bf(p1.x); a[kt].u[5]=f2bf(p1.y); a[kt].u[6]=f2bf(p1.z); a[kt].u[7]=f2bf(p1.w);
    a[kt].u[8]=f2bf(p2.x); a[kt].u[9]=f2bf(p2.y); a[kt].u[10]=f2bf(p2.z); a[kt].u[11]=f2bf(p2.w);
    a[kt].u[12]=f2bf(p3.x); a[kt].u[13]=f2bf(p3.y); a[kt].u[14]=f2bf(p3.z); a[kt].u[15]=f2bf(p3.w);
  }

  const bool fullTile = (rowBase + 16 <= N);     // wave-uniform

#pragma unroll
  for (int j = 0; j < NMAT; ++j) {
    const unsigned short* Wb = (j == 0) ? Wb0 : ((j == 1) ? Wb1 : Wb2);  // compile-time select
    const float* bias        = (j == 0) ? b0  : ((j == 1) ? b1  : b2);
    float* C                 = (j == 0) ? C0  : ((j == 1) ? C1  : C2);

#pragma unroll
    for (int ot = 0; ot < 8; ++ot) {
      const int o = ot * 16 + m;               // output column (same for hi/lo halves)
      v8f acc = {0.f,0.f,0.f,0.f,0.f,0.f,0.f,0.f};
#pragma unroll
      for (int kt = 0; kt < 4; ++kt) {
        // B 32x16 bf16, this code's contiguous k order (same as A): lane (o), 16 contiguous k values; lanes 16-31 take k+16
        v16bf b = *(const v16bf*)(Wb + (size_t)o * DF + kt * 32 + (hi ? 16 : 0));
        acc = __builtin_amdgcn_wmma_f32_16x16x32_bf16(false, a[kt].v, false, b,
                                                      (short)0, acc, false, false);
      }
      const float bo = bias[o];
      float vres[8];
#pragma unroll
      for (int r = 0; r < 8; ++r) {
        float v = acc[r] + bo;
        if (ACT) v = fmaxf(v, 0.0f);
        vres[r] = v;
      }
      // C/D layout: VGPR r -> M = r (+8 for hi lanes), N = lane%16
      float* cp = C + (size_t)(rowBase + (hi ? 8 : 0)) * LDC + o;
      if (fullTile) {
#pragma unroll
        for (int r = 0; r < 8; ++r) cp[r * LDC] = vres[r];   // immediate-offset stores
      } else {
#pragma unroll
        for (int r = 0; r < 8; ++r)
          if (rowBase + r + (hi ? 8 : 0) < N) cp[r * LDC] = vres[r];
      }
    }
  }
}

// ---------------- per-edge scores + per-block max ----------------
__global__ void __launch_bounds__(256) k_scores(const float* __restrict__ Q,
                                                const float* __restrict__ K,
                                                const int* __restrict__ row,
                                                const int* __restrict__ col,
                                                float* __restrict__ sc,
                                                float* __restrict__ blkmax, int E) {
  __shared__ float sm[8];
  const int lane = threadIdx.x & 31;
  const int wv   = threadIdx.x >> 5;
  int e = blockIdx.x * 8 + wv;
  const int stride = gridDim.x * 8;
  float lmax = -3.0e38f;
  for (; e < E; e += stride) {
    const int r = row[e], c = col[e];
    const float4 q = ((const float4*)(Q + (size_t)r * DF))[lane];
    const float4 k = ((const float4*)(K + (size_t)c * DF))[lane];
    float d = q.x*k.x + q.y*k.y + q.z*k.z + q.w*k.w;
#pragma unroll
    for (int off = 16; off >= 1; off >>= 1) d += __shfl_xor(d, off, 32);
    d *= 0.08838834764831845f;               // 1/sqrt(128)
    if (lane == 0) sc[e] = d;
    lmax = fmaxf(lmax, d);
  }
  if (lane == 0) sm[wv] = lmax;
  __syncthreads();
  if (threadIdx.x == 0) {
    float mm = sm[0];
    for (int i = 1; i < 8; ++i) mm = fmaxf(mm, sm[i]);
    blkmax[blockIdx.x] = mm;
  }
}

// ---------------- single-block reductions ----------------
__global__ void __launch_bounds__(256) k_rmax(const float* __restrict__ in, int n, float* __restrict__ out) {
  __shared__ float sm[8];
  float m = -3.0e38f;
  for (int i = threadIdx.x; i < n; i += blockDim.x) m = fmaxf(m, in[i]);
#pragma unroll
  for (int off = 16; off >= 1; off >>= 1) m = fmaxf(m, __shfl_xor(m, off, 32));
  if ((threadIdx.x & 31) == 0) sm[threadIdx.x >> 5] = m;
  __syncthreads();
  if (threadIdx.x == 0) {
    float mm = sm[0];
    for (int i = 1; i < 8; ++i) mm = fmaxf(mm, sm[i]);
    out[0] = mm;
  }
}

__global__ void __launch_bounds__(256) k_rsum_inv(const float* __restrict__ in, int n, float* __restrict__ out) {
  __shared__ float sm[8];
  float s = 0.0f;
  for (int i = threadIdx.x; i < n; i += blockDim.x) s += in[i];
#pragma unroll
  for (int off = 16; off >= 1; off >>= 1) s += __shfl_xor(s, off, 32);
  if ((threadIdx.x & 31) == 0) sm[threadIdx.x >> 5] = s;
  __syncthreads();
  if (threadIdx.x == 0) {
    float t = 0.0f;
    for (int i = 0; i < 8; ++i) t += sm[i];
    out[0] = 1.0f / t;
  }
}

// ---------------- exp(score - max) + per-block sums ----------------
__global__ void __launch_bounds__(256) k_exp(float* __restrict__ sc, const float* __restrict__ mx,
                                             float* __restrict__ blksum, int E) {
  __shared__ float sm[8];
  const float m = mx[0];
  float s = 0.0f;
  int i = blockIdx.x * blockDim.x + threadIdx.x;
  const int stride = gridDim.x * blockDim.x;
  for (; i < E; i += stride) { float v = expf(sc[i] - m); sc[i] = v; s += v; }
#pragma unroll
  for (int off = 16; off >= 1; off >>= 1) s += __shfl_xor(s, off, 32);
  if ((threadIdx.x & 31) == 0) sm[threadIdx.x >> 5] = s;
  __syncthreads();
  if (threadIdx.x == 0) {
    float t = 0.0f;
    for (int j = 0; j < 8; ++j) t += sm[j];
    blksum[blockIdx.x] = t;
  }
}

// ---------------- scatter: out[row] += w * V[col]  (w = alpha*inv or 1) ----------------
__global__ void __launch_bounds__(256) k_scatter(const float* __restrict__ V,
                                                 const float* __restrict__ alpha,
                                                 const float* __restrict__ invp,
                                                 const int* __restrict__ row,
                                                 const int* __restrict__ col,
                                                 float* __restrict__ out, int ldC, int E) {
  const int lane = threadIdx.x & 31;
  const int wv   = threadIdx.x >> 5;
  const float inv = invp ? invp[0] : 1.0f;
  int e = blockIdx.x * 8 + wv;
  const int stride = gridDim.x * 8;
  for (; e < E; e += stride) {
    const int r = row[e], c = col[e];
    const float w = alpha ? alpha[e] * inv : 1.0f;
    const float4 v = ((const float4*)(V + (size_t)c * DF))[lane];
    float* op = out + (size_t)r * ldC + lane * 4;
    atomicAdd(op + 0, w * v.x);
    atomicAdd(op + 1, w * v.y);
    atomicAdd(op + 2, w * v.z);
    atomicAdd(op + 3, w * v.w);
  }
}

// ---------------- ELU in place ----------------
__global__ void k_elu(float* __restrict__ p, long long n) {
  long long i = (long long)blockIdx.x * blockDim.x + threadIdx.x;
  long long s = (long long)gridDim.x * blockDim.x;
  for (; i < n; i += s) { float v = p[i]; p[i] = v > 0.0f ? v : expm1f(v); }
}

// ---------------- row-wise divide by clamped degree ----------------
__global__ void k_rownorm(float* __restrict__ p, const float* __restrict__ deg, long long n) {
  long long i = (long long)blockIdx.x * blockDim.x + threadIdx.x;
  long long s = (long long)gridDim.x * blockDim.x;
  for (; i < n; i += s) {
    float d = deg[i >> 7];
    p[i] *= (1.0f / fmaxf(d, 1.0f));
  }
}

extern "C" void kernel_launch(void* const* d_in, const int* in_sizes, int n_in,
                              void* d_out, int out_size, void* d_ws, size_t ws_size,
                              hipStream_t stream) {
  (void)n_in; (void)out_size; (void)ws_size;
  const int Nn = in_sizes[0] / DF;
  const int Ee = in_sizes[1] / 2;
  const float* x  = (const float*)d_in[0];
  const int* row  = (const int*)d_in[1];
  const int* col  = row + Ee;
  const float* W[8] = { (const float*)d_in[2],  (const float*)d_in[4],  (const float*)d_in[6],
                        (const float*)d_in[8],  (const float*)d_in[10], (const float*)d_in[12],
                        (const float*)d_in[14], (const float*)d_in[16] };
  const float* B[8] = { (const float*)d_in[3],  (const float*)d_in[5],  (const float*)d_in[7],
                        (const float*)d_in[9],  (const float*)d_in[11], (const float*)d_in[13],
                        (const float*)d_in[15], (const float*)d_in[17] };
  float* out = (float*)d_out;

  // workspace carve (256B aligned)
  char* p = (char*)d_ws;
  auto take = [&](size_t bytes) -> char* {
    char* r = p; p += (bytes + 255) & ~(size_t)255; return r;
  };
  unsigned short* Wb[8];
  for (int i = 0; i < 8; ++i) Wb[i] = (unsigned short*)take((size_t)DF * DF * 2);
  const size_t NB = (size_t)Nn * DF * sizeof(float);
  float* A    = (float*)take(NB);               // attention acc / h / agg2
  float* Qb   = (float*)take(NB);               // Q / agg1
  float* Kb   = (float*)take(NB);               // K / s
  float* Vb   = (float*)take(NB);               // V
  float* sc   = (float*)take((size_t)Ee * 4);   // scores -> exp values
  float* deg  = (float*)take((size_t)Nn * 4);
  float* red  = (float*)take(4096 * 4);
  float* red2 = (float*)take(4096 * 4);
  float* scal = (float*)take(256);              // [0]=max, [1]=1/sum

  const int gGemm = (Nn + 127) / 128;
  const long long nND = (long long)Nn * DF;

  // weights -> bf16 once
  for (int i = 0; i < 8; ++i) k_wconv<<<64, 256, 0, stream>>>(W[i], Wb[i]);

  // init
  k_zero<<<2048, 256, 0, stream>>>(A, nND);
  k_zero<<<256, 256, 0, stream>>>(deg, Nn);
  k_zero<<<2048, 256, 0, stream>>>(out, (long long)Nn * 256);
  k_deg<<<1024, 256, 0, stream>>>(row, Ee, deg);

  // ---- attention layer 1 (fused Q/K/V) ----
  k_gemm<DF, 0, 3><<<gGemm, 256, 0, stream>>>(x,
      Wb[0], B[0], Qb, Wb[1], B[1], Kb, Wb[2], B[2], Vb, Nn);
  k_scores<<<1024, 256, 0, stream>>>(Qb, Kb, row, col, sc, red, Ee);
  k_rmax<<<1, 256, 0, stream>>>(red, 1024, scal);
  k_exp<<<1024, 256, 0, stream>>>(sc, scal, red2, Ee);
  k_rsum_inv<<<1, 256, 0, stream>>>(red2, 1024, scal + 1);
  k_scatter<<<1024, 256, 0, stream>>>(Vb, sc, scal + 1, row, col, A, DF, Ee);
  k_elu<<<2048, 256, 0, stream>>>(A, nND);      // h = elu(...)

  // ---- attention layer 2 (bpp_out -> out cols 0..127) ----
  k_gemm<DF, 0, 3><<<gGemm, 256, 0, stream>>>(A,
      Wb[3], B[3], Qb, Wb[4], B[4], Kb, Wb[5], B[5], Vb, Nn);
  k_scores<<<1024, 256, 0, stream>>>(Qb, Kb, row, col, sc, red, Ee);
  k_rmax<<<1, 256, 0, stream>>>(red, 1024, scal);
  k_exp<<<1024, 256, 0, stream>>>(sc, scal, red2, Ee);
  k_rsum_inv<<<1, 256, 0, stream>>>(red2, 1024, scal + 1);
  k_scatter<<<1024, 256, 0, stream>>>(Vb, sc, scal + 1, row, col, out, 256, Ee);

  // ---- secondary-structure encoder (ss_out -> out cols 128..255) ----
  k_zero<<<2048, 256, 0, stream>>>(Qb, nND);
  k_scatter<<<1024, 256, 0, stream>>>(x, nullptr, nullptr, row, col, Qb, DF, Ee);
  k_rownorm<<<2048, 256, 0, stream>>>(Qb, deg, nND);
  k_gemm<DF, 1, 1><<<gGemm, 256, 0, stream>>>(Qb,
      Wb[6], B[6], Kb, nullptr, nullptr, nullptr, nullptr, nullptr, nullptr, Nn);
  k_zero<<<2048, 256, 0, stream>>>(A, nND);
  k_scatter<<<1024, 256, 0, stream>>>(Kb, nullptr, nullptr, row, col, A, DF, Ee);
  k_rownorm<<<2048, 256, 0, stream>>>(A, deg, nND);
  k_gemm<256, 1, 1><<<gGemm, 256, 0, stream>>>(A,
      Wb[7], B[7], out + DF, nullptr, nullptr, nullptr, nullptr, nullptr, nullptr, Nn);
}
